// Attention_25211458027979
// MI455X (gfx1250) — matrix operand order repaired and checked by reading, not compiled, not run on hardware
//
#include <hip/hip_runtime.h>
#include <hip/hip_bf16.h>
#include <math.h>

#define DIM   1024
#define NHEAD 16
#define HDIM  64
#define BATCH 2
#define SEQ   2048
#define MTOT  (BATCH * SEQ)   // 4096 rows

typedef __attribute__((ext_vector_type(2)))  float        v2f;
typedef __attribute__((ext_vector_type(8)))  float        v8f;
typedef __attribute__((ext_vector_type(4)))  int          v4i;
typedef __attribute__((ext_vector_type(4)))  unsigned int v4u;
typedef __attribute__((ext_vector_type(8)))  int          v8i;
typedef __attribute__((ext_vector_type(8)))  __bf16       v8bf;
typedef __attribute__((ext_vector_type(16))) __bf16       v16bf;

#if __has_builtin(__builtin_amdgcn_global_load_async_to_lds_b128)
#define HAVE_ASYNC_LDS 1
#else
#define HAVE_ASYNC_LDS 0
#endif

#if __has_builtin(__builtin_amdgcn_tensor_load_to_lds)
#define HAVE_TDM 1
#else
#define HAVE_TDM 0
#endif

// Probe-verified: async builtin takes (v4i addrspace(1)*, v4i addrspace(3)*, imm, imm)
typedef __attribute__((address_space(1))) v4i* gv4ip;
typedef __attribute__((address_space(3))) v4i* lv4ip;

// Per-lane async 16-byte global -> LDS copy (ASYNCcnt tracked). Fallback: sync copy.
static __device__ __forceinline__ void async_g2l_16B(const void* g, void* l) {
#if HAVE_ASYNC_LDS
  __builtin_amdgcn_global_load_async_to_lds_b128((gv4ip)g, (lv4ip)l, 0, 0);
#else
  *(float4*)l = *(const float4*)g;
#endif
}
static __device__ __forceinline__ void wait_async() {
#if HAVE_ASYNC_LDS
#if __has_builtin(__builtin_amdgcn_s_wait_asynccnt)
  __builtin_amdgcn_s_wait_asynccnt(0);
#else
  asm volatile("s_wait_asynccnt 0x0" ::: "memory");
#endif
#endif
}
static __device__ __forceinline__ void wait_tensor() {
#if __has_builtin(__builtin_amdgcn_s_wait_tensorcnt)
  __builtin_amdgcn_s_wait_tensorcnt(0);
#else
  asm volatile("s_wait_tensorcnt 0x0" ::: "memory");
#endif
}

// LDS byte offset of a __shared__ object (AS3 pointers are LDS byte offsets).
static __device__ __forceinline__ unsigned lds_off(const void* p) {
  return (unsigned)(size_t)(__attribute__((address_space(3))) const char*)p;
}

#if HAVE_TDM
// Issue one TDM 2D tile load: tile (tile0 x tile1) elements of size (1<<dsz) bytes
// from a row-major tensor (dim0 x dim1, row stride stride0 elements) into LDS
// at byte offset ldsAddr. D# layout per CDNA5 ISA 8.3-8.5; groups 2/3 zero (2D).
// This toolchain's builtin is the 6-arg form: (g0, g1, g2, g3, g4, cpol).
static __device__ __forceinline__ void tdm_load_2d(
    unsigned ldsAddr, const void* gptr,
    unsigned dim0, unsigned dim1, unsigned tile0, unsigned tile1,
    unsigned stride0, unsigned dsz)
{
  unsigned long long ga = (unsigned long long)(size_t)gptr;
  v4u g0;
  g0[0] = 1u;                                   // count=1, user descriptor
  g0[1] = ldsAddr;                              // lds_addr (bytes)
  g0[2] = (unsigned)(ga & 0xFFFFFFFFu);         // global_addr[31:0]
  g0[3] = (unsigned)((ga >> 32) & 0x01FFFFFFu)  // global_addr[56:32]
          | (2u << 30);                         // type=2 ("image")
  v8i g1;
  g1[0] = (int)(dsz << 16);                     // data_size
  g1[1] = (int)((dim0 & 0xFFFFu) << 16);        // tensor_dim0[15:0]
  g1[2] = (int)((dim0 >> 16) | ((dim1 & 0xFFFFu) << 16));   // dim0[31:16], dim1[15:0]
  g1[3] = (int)((dim1 >> 16) | (tile0 << 16));  // dim1[31:16], tile_dim0
  g1[4] = (int)tile1;                           // tile_dim1 (tile_dim2=0)
  g1[5] = (int)stride0;                         // tensor_dim0_stride[31:0]
  g1[6] = 0;                                    // stride hi, dim1_stride lo
  g1[7] = 0;
  v4i gz4 = {0, 0, 0, 0};
  v8i gz8 = {0, 0, 0, 0, 0, 0, 0, 0};
  __builtin_amdgcn_tensor_load_to_lds(g0, g1, gz4, gz4, gz8, 0);
}
#endif

static __device__ __forceinline__ v16bf cat8(v8bf lo, v8bf hi) {
  return __builtin_shufflevector(lo, hi, 0,1,2,3,4,5,6,7,8,9,10,11,12,13,14,15);
}
static __device__ __forceinline__ v8f vzero8() {
  v8f z;
#pragma unroll
  for (int i = 0; i < 8; ++i) z[i] = 0.0f;
  return z;
}

// ---------------------------------------------------------------------------
// Kernel 1/4: fp32 GEMM  C[M,N] = A[M,K] * B[K,N]  via V_WMMA_F32_16X16X4_F32.
// 8 waves, BM=128/BN=64/BK=32, double-buffered LDS:
//   A tile staged by one TDM tensor_load_to_lds per tile (wave 0 issues,
//   TENSORcnt-waited; async-b128 fallback), B tile staged transposed [col][k]
//   (pad stride 34) so each WMMA B-fragment pair is one contiguous ds_load_b64.
// ---------------------------------------------------------------------------
__global__ __launch_bounds__(256)
void gemm_f32_wmma_k(const float* __restrict__ A, const float* __restrict__ B,
                     float* __restrict__ C, int M, int N, int K)
{
  constexpr int BM = 128, BN = 64, BK = 32, BKP = BK + 2;
  __shared__ __align__(16) float As[2][BM * BK];
  __shared__ __align__(16) float Bst[2][BN * BKP];

  const int lane = threadIdx.x & 31;
  const int wv   = threadIdx.x >> 5;
  const int g    = lane >> 4;
  const int lc   = lane & 15;

  const int tilesN = N / BN;
  const int tm = (blockIdx.x / tilesN) * BM;
  const int tn = (blockIdx.x % tilesN) * BN;

  v8f acc[4];
#pragma unroll
  for (int t = 0; t < 4; ++t) acc[t] = vzero8();

  auto stage = [&](int buf, int k0) {
    // A tile BMxBK: one TDM DMA issued by wave 0 (fallback: per-lane async b128).
#if HAVE_TDM
    if (wv == 0)
      tdm_load_2d(lds_off(&As[buf][0]), &A[(size_t)tm * K + k0],
                  (unsigned)K, (unsigned)M, BK, BM, (unsigned)K, /*4B*/2);
#else
    for (int i = threadIdx.x * 4; i < BM * BK; i += 1024) {
      int r = i >> 5, c = i & 31;
      async_g2l_16B(&A[(size_t)(tm + r) * K + k0 + c], &As[buf][i]);
    }
#endif
    // B tile BKxBN, transposed into [col][k] with padded stride.
    for (int i = threadIdx.x * 4; i < BK * BN; i += 1024) {
      int r = i >> 6, c = i & 63;
      float4 v = *(const float4*)&B[(size_t)(k0 + r) * N + tn + c];
      float* bp = &Bst[buf][0];
      bp[(c + 0) * BKP + r] = v.x;
      bp[(c + 1) * BKP + r] = v.y;
      bp[(c + 2) * BKP + r] = v.z;
      bp[(c + 3) * BKP + r] = v.w;
    }
  };

  auto stage_wait = [&]() {
#if HAVE_TDM
    if (wv == 0) wait_tensor();
#else
    wait_async();
#endif
  };

  stage(0, 0);
  stage_wait();
  __syncthreads();

  int buf = 0;
  for (int k0 = 0; k0 < K; k0 += BK) {
    if (k0 + BK < K) stage(buf ^ 1, k0 + BK);  // overlap next-tile staging

    const float* arow = &As[buf][(wv * 16 + lc) * BK];
    const float* b0 = &Bst[buf][(0  + lc) * BKP];
    const float* b1 = &Bst[buf][(16 + lc) * BKP];
    const float* b2 = &Bst[buf][(32 + lc) * BKP];
    const float* b3 = &Bst[buf][(48 + lc) * BKP];
#pragma unroll
    for (int kk = 0; kk < BK; kk += 4) {
      v2f a = *(const v2f*)(arow + kk + 2 * g);
      acc[0] = __builtin_amdgcn_wmma_f32_16x16x4_f32(
          false, a, false, *(const v2f*)(b0 + kk + 2 * g), (short)0, acc[0], false, false);
      acc[1] = __builtin_amdgcn_wmma_f32_16x16x4_f32(
          false, a, false, *(const v2f*)(b1 + kk + 2 * g), (short)0, acc[1], false, false);
      acc[2] = __builtin_amdgcn_wmma_f32_16x16x4_f32(
          false, a, false, *(const v2f*)(b2 + kk + 2 * g), (short)0, acc[2], false, false);
      acc[3] = __builtin_amdgcn_wmma_f32_16x16x4_f32(
          false, a, false, *(const v2f*)(b3 + kk + 2 * g), (short)0, acc[3], false, false);
    }
    stage_wait();     // next tile's DMA landed in LDS
    __syncthreads();  // everyone done reading current tile + B stores visible
    buf ^= 1;
  }

#pragma unroll
  for (int t = 0; t < 4; ++t)
#pragma unroll
    for (int r = 0; r < 8; ++r)
      C[(size_t)(tm + wv * 16 + r + 8 * g) * N + tn + t * 16 + lc] = acc[t][r];
}

// ---------------------------------------------------------------------------
// Kernel 2: per-(b,n,head) RMSNorm on q/k + RoPE + cast q/k/v -> bf16.
// qkv layout in: row bn has [3][H][64] fp32. Out: [B*H][N][64] bf16.
// ---------------------------------------------------------------------------
__global__ __launch_bounds__(256)
void rmsnorm_rope_cast_k(const float* __restrict__ qkv,
                         const float* __restrict__ cosb, const float* __restrict__ sinb,
                         const float* __restrict__ wq, const float* __restrict__ wk,
                         __bf16* __restrict__ qb, __bf16* __restrict__ kb,
                         __bf16* __restrict__ vb)
{
  const int idx = blockIdx.x * blockDim.x + threadIdx.x;  // 0 .. B*N*H-1
  const int h  = idx & (NHEAD - 1);
  const int bn = idx >> 4;
  const int n  = bn & (SEQ - 1);
  const int b  = bn >> 11;

  const float* qr = qkv + (size_t)bn * (3 * DIM) + h * HDIM;
  const float* kr = qr + DIM;
  const float* vr = qr + 2 * DIM;

  float sq = 0.f, sk = 0.f;
#pragma unroll
  for (int d = 0; d < HDIM; d += 4) {
    float4 a = *(const float4*)(qr + d);
    float4 c = *(const float4*)(kr + d);
    sq += a.x * a.x + a.y * a.y + a.z * a.z + a.w * a.w;
    sk += c.x * c.x + c.y * c.y + c.z * c.z + c.w * c.w;
  }
  const float rq = rsqrtf(sq * (1.0f / HDIM) + 1e-6f);
  const float rk = rsqrtf(sk * (1.0f / HDIM) + 1e-6f);

  const size_t o = (((size_t)b * NHEAD + h) * SEQ + n) * HDIM;
  const float* cr = cosb + (size_t)n * HDIM;
  const float* sr = sinb + (size_t)n * HDIM;

  for (int d = 0; d < HDIM / 2; ++d) {
    float c0 = cr[d],      s0 = sr[d];
    float c1 = cr[d + 32], s1 = sr[d + 32];
    float q0 = qr[d] * rq * wq[d];
    float q1 = qr[d + 32] * rq * wq[d + 32];
    float k0 = kr[d] * rk * wk[d];
    float k1 = kr[d + 32] * rk * wk[d + 32];
    qb[o + d]      = (__bf16)(q0 * c0 - q1 * s0);
    qb[o + d + 32] = (__bf16)(q1 * c1 + q0 * s1);
    kb[o + d]      = (__bf16)(k0 * c0 - k1 * s0);
    kb[o + d + 32] = (__bf16)(k1 * c1 + k0 * s1);
    vb[o + d]      = (__bf16)vr[d];
    vb[o + d + 32] = (__bf16)vr[d + 32];
  }
}

// ---------------------------------------------------------------------------
// Kernel 3: flash attention per (b,h, 128-query block), bf16 WMMA, fp32 acc.
// 8 waves x 16 query rows. Double-buffered K/V staging:
//   K block (64x64, row-major as consumed) via async b128 global->LDS,
//   V block transposed [d][key] in LDS so WMMA B-fragments are contiguous.
// P goes C-layout -> wave-private LDS -> A-layout (per-wave in-order LDS).
// ---------------------------------------------------------------------------
__global__ __launch_bounds__(256)
void flash_attn_k(const __bf16* __restrict__ Q, const __bf16* __restrict__ Kb,
                  const __bf16* __restrict__ Vb, float* __restrict__ out)
{
  __shared__ __align__(16) __bf16 Ks[2][64 * 64];     // [key][d]
  __shared__ __align__(16) __bf16 Vt[2][64 * 64];     // [d][key]
  __shared__ __align__(16) __bf16 Ps[8 * 16 * 64];    // per-wave 16x64 P tile

  const int lane = threadIdx.x & 31;
  const int wv   = threadIdx.x >> 5;
  const int g    = lane >> 4;
  const int lc   = lane & 15;

  const int qblocks = SEQ / 128;
  const int bh = blockIdx.x / qblocks;
  const int qt = blockIdx.x % qblocks;
  const int b  = bh >> 4, h = bh & 15;
  const size_t base = (size_t)bh * SEQ * HDIM;
  const int q0r = qt * 128 + wv * 16;

  // Q A-fragments (16x32 bf16): per lane two contiguous 8-elem runs:
  // K = f*32 + 8g + [0,8) and f*32 + 16 + 8g + [0,8).
  v16bf qa[2];
  {
    const __bf16* qp = Q + base + (size_t)(q0r + lc) * HDIM;
#pragma unroll
    for (int f = 0; f < 2; ++f)
      qa[f] = cat8(*(const v8bf*)(qp + f * 32 + 8 * g),
                   *(const v8bf*)(qp + f * 32 + 16 + 8 * g));
  }

  float m[8], l[8];
  v8f o[4];
#pragma unroll
  for (int r = 0; r < 8; ++r) { m[r] = -1e30f; l[r] = 0.f; }
#pragma unroll
  for (int t = 0; t < 4; ++t) o[t] = vzero8();

  __bf16* pw = &Ps[wv * 16 * 64];

  auto stageKV = [&](int buf, int kb0) {
    for (int i = threadIdx.x * 8; i < 64 * 64; i += 2048) {
      int key = i >> 6, d = i & 63;
      async_g2l_16B(Kb + base + (size_t)(kb0 + key) * HDIM + d, &Ks[buf][i]);
      v8bf vv = *(const v8bf*)(Vb + base + (size_t)(kb0 + key) * HDIM + d);
#pragma unroll
      for (int e = 0; e < 8; ++e) Vt[buf][(d + e) * 64 + key] = vv[e];
    }
  };

  stageKV(0, 0);
  wait_async();
  __syncthreads();

  int buf = 0;
  for (int kb0 = 0; kb0 < SEQ; kb0 += 64) {
    if (kb0 + 64 < SEQ) stageKV(buf ^ 1, kb0 + 64);  // overlap next block

    // S = Q x K^T : 4 column tiles of 16 keys, 2 k-steps over d.
    v8f s[4];
#pragma unroll
    for (int t = 0; t < 4; ++t) {
      v8f accs = vzero8();
#pragma unroll
      for (int f = 0; f < 2; ++f) {
        const __bf16* kp = &Ks[buf][(16 * t + lc) * 64 + f * 32 + 8 * g];
        v16bf bfrag = cat8(*(const v8bf*)kp, *(const v8bf*)(kp + 16));
        accs = __builtin_amdgcn_wmma_f32_16x16x32_bf16(
            false, qa[f], false, bfrag, (short)0, accs, false, false);
      }
#pragma unroll
      for (int r = 0; r < 8; ++r) accs[r] *= 0.125f;  // 1/sqrt(64)
      s[t] = accs;
    }

    // Online softmax: row = r + 8g, full row lives across 16 lanes.
#pragma unroll
    for (int r = 0; r < 8; ++r) {
      float mx = fmaxf(fmaxf(s[0][r], s[1][r]), fmaxf(s[2][r], s[3][r]));
#pragma unroll
      for (int msk = 1; msk < 16; msk <<= 1)
        mx = fmaxf(mx, __shfl_xor(mx, msk, 32));
      float mn = fmaxf(m[r], mx);
      float alpha = __expf(m[r] - mn);
      m[r] = mn;
      l[r] *= alpha;
#pragma unroll
      for (int t = 0; t < 4; ++t) o[t][r] *= alpha;
    }

    // P = exp(S - m): keep for row sums, spill to wave-private LDS for relayout.
#pragma unroll
    for (int t = 0; t < 4; ++t)
#pragma unroll
      for (int r = 0; r < 8; ++r) {
        float p = __expf(s[t][r] - m[r]);
        s[t][r] = p;
        pw[(r + 8 * g) * 64 + 16 * t + lc] = (__bf16)p;
      }
#pragma unroll
    for (int r = 0; r < 8; ++r) {
      float sum = s[0][r] + s[1][r] + s[2][r] + s[3][r];
#pragma unroll
      for (int msk = 1; msk < 16; msk <<= 1)
        sum += __shfl_xor(sum, msk, 32);
      l[r] += sum;
    }

    // O += P x V : A = P (A-layout from LDS), B = V via transposed LDS tile.
#pragma unroll
    for (int f = 0; f < 2; ++f) {
      const __bf16* pp = pw + lc * 64 + f * 32;
      v16bf pa = cat8(*(const v8bf*)(pp + 8 * g),
                      *(const v8bf*)(pp + 16 + 8 * g));
#pragma unroll
      for (int c = 0; c < 4; ++c) {
        const __bf16* vp = &Vt[buf][(16 * c + lc) * 64 + f * 32 + 8 * g];
        v16bf vfrag = cat8(*(const v8bf*)vp, *(const v8bf*)(vp + 16));
        o[c] = __builtin_amdgcn_wmma_f32_16x16x32_bf16(
            false, pa, false, vfrag, (short)0, o[c], false, false);
      }
    }

    wait_async();
    __syncthreads();
    buf ^= 1;
  }

  // Normalize and write attn in [B, N, H*D] fp32 (ready for out-projection GEMM).
#pragma unroll
  for (int c = 0; c < 4; ++c)
#pragma unroll
    for (int r = 0; r < 8; ++r) {
      int row = q0r + r + 8 * g;
      out[((size_t)(b * SEQ + row)) * DIM + h * HDIM + 16 * c + lc] =
          o[c][r] / l[r];
    }
}

// ---------------------------------------------------------------------------
extern "C" void kernel_launch(void* const* d_in, const int* in_sizes, int n_in,
                              void* d_out, int out_size, void* d_ws, size_t ws_size,
                              hipStream_t stream)
{
  const float* hs   = (const float*)d_in[0];   // hidden_states (2,2048,1024)
  const float* cosb = (const float*)d_in[1];   // cos (2048,64)
  const float* sinb = (const float*)d_in[2];   // sin (2048,64)
  const float* wqkv = (const float*)d_in[3];   // (1024,3072)
  const float* nqw  = (const float*)d_in[4];   // (64,)
  const float* nkw  = (const float*)d_in[5];   // (64,)
  const float* wout = (const float*)d_in[6];   // (1024,1024)
  float* out = (float*)d_out;

  char* ws = (char*)d_ws;
  float*  qkv  = (float*)ws;                               // 4096*3072 f32 (48 MB)
  size_t  qkvBytes = (size_t)MTOT * 3 * DIM * sizeof(float);
  __bf16* qb   = (__bf16*)(ws + qkvBytes);                 // 3 x 8 MB bf16
  __bf16* kbuf = qb + (size_t)MTOT * DIM;
  __bf16* vbuf = kbuf + (size_t)MTOT * DIM;
  float*  attn = (float*)ws;                               // reuse qkv region

  // 1) QKV projection (fp32 WMMA): [4096,1024] x [1024,3072]
  gemm_f32_wmma_k<<<(MTOT / 128) * (3 * DIM / 64), 256, 0, stream>>>(
      hs, wqkv, qkv, MTOT, 3 * DIM, DIM);

  // 2) RMSNorm + RoPE + bf16 cast -> q/k/v in [B*H][N][64]
  rmsnorm_rope_cast_k<<<(MTOT * NHEAD) / 256, 256, 0, stream>>>(
      qkv, cosb, sinb, nqw, nkw, qb, kbuf, vbuf);

  // 3) Flash attention (bf16 WMMA, fp32 acc) -> attn [4096,1024] fp32
  flash_attn_k<<<BATCH * NHEAD * (SEQ / 128), 256, 0, stream>>>(
      qb, kbuf, vbuf, attn);

  // 4) Output projection (fp32 WMMA): [4096,1024] x [1024,1024]
  gemm_f32_wmma_k<<<(MTOT / 128) * (DIM / 64), 256, 0, stream>>>(
      attn, wout, out, MTOT, DIM, DIM);
}
